// DNA_GNN_77524159693152
// MI455X (gfx1250) — matrix operand order repaired and checked by reading, not compiled, not run on hardware
//
#include <hip/hip_runtime.h>
#include <hip/hip_bf16.h>
#include <math.h>

// ---------------------------------------------------------------------------
// GCN forward for MI455X (gfx1250, wave32).
//   x[N,4] -> GEMM1 -> agg(norm) -> +b1,relu -> GEMM2 (WMMA bf16 split) ->
//   agg(norm) -> +b2,relu -> mean pool -> sigmoid(dot(Wl)+bl)
// ---------------------------------------------------------------------------

#define HID  256
#define H2   128

typedef __attribute__((ext_vector_type(16))) __bf16 v16bf;
typedef __attribute__((ext_vector_type(8)))  float  v8f;

static __device__ __forceinline__ unsigned short f32_to_bf16(float f) {
  unsigned int u = __float_as_uint(f);
  u += 0x7FFFu + ((u >> 16) & 1u);      // round-to-nearest-even
  return (unsigned short)(u >> 16);
}
static __device__ __forceinline__ float bf16_to_f32(unsigned short h) {
  return __uint_as_float(((unsigned int)h) << 16);
}

// ---------------- degree / norm ----------------
__global__ void k_deg_init(float* deg, int N) {
  int i = blockIdx.x * blockDim.x + threadIdx.x;
  if (i < N) deg[i] = 1.0f;                       // self-loop
}
__global__ void k_deg_edges(const int* __restrict__ col, float* deg, int E) {
  int e = blockIdx.x * blockDim.x + threadIdx.x;
  if (e < E) atomicAdd(&deg[col[e]], 1.0f);
}
__global__ void k_dis(float* deg, int N) {
  int i = blockIdx.x * blockDim.x + threadIdx.x;
  if (i < N) deg[i] = rsqrtf(deg[i]);             // deg >= 1 always
}

// ---------------- W2 pack: [K=256][N=128] f32 -> transposed bf16 hi/lo -----
__global__ void k_w2pack(const float* __restrict__ W2,
                         unsigned short* __restrict__ w2hi,
                         unsigned short* __restrict__ w2lo) {
  int idx = blockIdx.x * blockDim.x + threadIdx.x;   // < 256*128
  if (idx >= HID * H2) return;
  int k = idx / H2, n = idx % H2;
  float v = W2[idx];
  unsigned short hi = f32_to_bf16(v);
  float rem = v - bf16_to_f32(hi);
  w2hi[n * HID + k] = hi;
  w2lo[n * HID + k] = f32_to_bf16(rem);
}

// ---------------- GEMM1: t1 = x @ W1  (K=4, memory bound) ------------------
__global__ void k_gemm1(const float* __restrict__ x, const float* __restrict__ W1,
                        float* __restrict__ t1, int N) {
  __shared__ float w[4 * HID];
  for (int i = threadIdx.x; i < 4 * HID; i += blockDim.x) w[i] = W1[i];
  __syncthreads();
  int node = blockIdx.x;
  if (node >= N) return;
  float x0 = x[node * 4 + 0], x1 = x[node * 4 + 1];
  float x2 = x[node * 4 + 2], x3 = x[node * 4 + 3];
  int j = threadIdx.x;                               // 256 threads = 256 cols
  t1[(size_t)node * HID + j] =
      x0 * w[j] + x1 * w[HID + j] + x2 * w[2 * HID + j] + x3 * w[3 * HID + j];
}

// ---------------- aggregation: dst[col] += src[row] * norm -----------------
// One thread per (edge, 4 features). e in [E, E+N) are self-loops.
template <int W>
__global__ void k_agg(const float* __restrict__ src, float* __restrict__ dst,
                      const int* __restrict__ row, const int* __restrict__ col,
                      const float* __restrict__ dis, int E, int N) {
  constexpr int TPE = W / 4;                         // threads per edge
  long long idx = (long long)blockIdx.x * blockDim.x + threadIdx.x;
  long long e = idx / TPE;
  int q = (int)(idx % TPE);
  if (e >= (long long)E + N) return;
  int r, c; float nrm;
  if (e < E) {
    r = row[e]; c = col[e];
    nrm = dis[r] * dis[c];
  } else {
    r = c = (int)(e - E);
    float d = dis[r]; nrm = d * d;
  }
  const float4 v = *(const float4*)(src + (size_t)r * W + q * 4);
  float* d = dst + (size_t)c * W + q * 4;
  atomicAdd(d + 0, v.x * nrm);
  atomicAdd(d + 1, v.y * nrm);
  atomicAdd(d + 2, v.z * nrm);
  atomicAdd(d + 3, v.w * nrm);
}

// ---------------- bias + relu (in place) -----------------------------------
template <int W>
__global__ void k_bias_relu(float* __restrict__ buf, const float* __restrict__ b,
                            long long total) {
  long long idx = (long long)blockIdx.x * blockDim.x + threadIdx.x;
  if (idx >= total) return;
  int j = (int)(idx % W);
  float v = buf[idx] + b[j];
  buf[idx] = fmaxf(v, 0.0f);
}

// ---------------- GEMM2: t2 = h1 @ W2 via WMMA bf16 (split hi/lo) ----------
// Block = 256 threads = 8 waves. Block covers 16 rows x 128 cols; wave w does
// the 16x16 tile at cols [16w, 16w+16). K = 256 in 8 steps of 32.
// Accuracy: A,B split into bf16 hi + bf16 residual; D = AhBh + AlBh + AhBl.
__global__ void __launch_bounds__(256) k_gemm2(
    const float* __restrict__ h1,
    const unsigned short* __restrict__ w2hi,
    const unsigned short* __restrict__ w2lo,
    float* __restrict__ t2, int N) {
  constexpr int AS = HID + 2;                        // padded LDS row stride
  __shared__ unsigned short a_hi[16 * AS];
  __shared__ unsigned short a_lo[16 * AS];
  const int tid = threadIdx.x;
  const int m0 = blockIdx.x * 16;

  // Stage A tile (16 x 256) into LDS as bf16 hi/lo.
  for (int t = tid; t < 16 * HID; t += 256) {
    int rr = t >> 8, kk = t & (HID - 1);
    int gm = m0 + rr;
    float v = (gm < N) ? h1[(size_t)gm * HID + kk] : 0.0f;
    unsigned short hi = f32_to_bf16(v);
    a_hi[rr * AS + kk] = hi;
    a_lo[rr * AS + kk] = f32_to_bf16(v - bf16_to_f32(hi));
  }
  __syncthreads();

  const int lane  = tid & 31;
  const int wave  = tid >> 5;
  const int n0    = wave * 16;
  const int nn    = n0 + (lane & 15);
  const int mrow  = lane & 15;
  const int hiK   = (lane >> 4) * 8;                 // A: upper-half lanes +8
  const int hiKB  = (lane >> 4) * 16;                // unused; B loads use hiK (+8) like A

  v8f acc = {};
  for (int ks = 0; ks < 8; ++ks) {
    const int kbase = ks * 32;
    union { v16bf v; unsigned int u[8]; } Ah, Al, Bh, Bl;
#pragma unroll
    for (int g = 0; g < 8; ++g) {
      // A 16x32 bf16 layout (ISA 7.12.2): vgpr g -> K = 2*(g&3)+16*(g>>2)+8*(lane>>4)
      int ka = kbase + 2 * (g & 3) + 16 * (g >> 2) + hiK;
      // B 32x16 bf16 layout: vgpr g -> K = 2*(g&3)+16*(g>>2)+8*(lane>>4), col = lane&15
      int kb = kbase + 2 * (g & 3) + 16 * (g >> 2) + hiK;
      Ah.u[g] = *(const unsigned int*)(a_hi + mrow * AS + ka);
      Al.u[g] = *(const unsigned int*)(a_lo + mrow * AS + ka);
      Bh.u[g] = *(const unsigned int*)(w2hi + nn * HID + kb);
      Bl.u[g] = *(const unsigned int*)(w2lo + nn * HID + kb);
    }
    acc = __builtin_amdgcn_wmma_f32_16x16x32_bf16(false, Ah.v, false, Bh.v,
                                                  (short)0, acc, false, false);
    acc = __builtin_amdgcn_wmma_f32_16x16x32_bf16(false, Al.v, false, Bh.v,
                                                  (short)0, acc, false, false);
    acc = __builtin_amdgcn_wmma_f32_16x16x32_bf16(false, Ah.v, false, Bl.v,
                                                  (short)0, acc, false, false);
  }

  // D layout: vgpr v -> row m = v + 8*(lane>>4), col = lane&15.
  const int mbase = m0 + (lane >> 4) * 8;
#pragma unroll
  for (int v = 0; v < 8; ++v) {
    int gm = mbase + v;
    if (gm < N) t2[(size_t)gm * H2 + n0 + (lane & 15)] = acc[v];
  }
}

// ---------------- mean pool partials ---------------------------------------
__global__ void k_pool(const float* __restrict__ h2, float* __restrict__ gsum, int N) {
  int j  = threadIdx.x & (H2 - 1);
  int hh = threadIdx.x >> 7;                          // 0 or 1
  float s = 0.0f;
  for (long long i = (long long)blockIdx.x * 2 + hh; i < N; i += (long long)gridDim.x * 2)
    s += h2[i * H2 + j];
  __shared__ float tmp[256];
  tmp[threadIdx.x] = s;
  __syncthreads();
  if (hh == 0) atomicAdd(&gsum[j], tmp[threadIdx.x] + tmp[threadIdx.x + H2]);
}

// ---------------- head: sigmoid(mean @ Wl + bl) ----------------------------
__global__ void k_final(const float* __restrict__ gsum, const float* __restrict__ Wl,
                        const float* __restrict__ bl, float* __restrict__ out, int N) {
  __shared__ float red[H2];
  int j = threadIdx.x;
  red[j] = (gsum[j] / (float)N) * Wl[j];
  __syncthreads();
  for (int s = H2 / 2; s > 0; s >>= 1) {
    if (j < s) red[j] += red[j + s];
    __syncthreads();
  }
  if (j == 0) out[0] = 1.0f / (1.0f + expf(-(red[0] + bl[0])));
}

// ---------------------------------------------------------------------------
extern "C" void kernel_launch(void* const* d_in, const int* in_sizes, int n_in,
                              void* d_out, int out_size, void* d_ws, size_t ws_size,
                              hipStream_t stream) {
  (void)n_in; (void)out_size; (void)ws_size;
  const float* x   = (const float*)d_in[0];
  const int*   ei  = (const int*)d_in[1];
  const float* W1  = (const float*)d_in[2];
  const float* b1  = (const float*)d_in[3];
  const float* W2  = (const float*)d_in[4];
  const float* b2  = (const float*)d_in[5];
  const float* Wl  = (const float*)d_in[6];
  const float* bl  = (const float*)d_in[7];
  float* out = (float*)d_out;

  const int N = in_sizes[0] / 4;
  const int E = in_sizes[1] / 2;
  const int* row = ei;
  const int* col = ei + E;

  // workspace carve-out (256B aligned slices)
  char* base = (char*)d_ws;
  size_t off = 0;
  auto carve = [&](size_t bytes) -> void* {
    void* p = base + off;
    off += (bytes + 255) & ~(size_t)255;
    return p;
  };
  float*          deg   = (float*)carve((size_t)N * 4);
  float*          bufA  = (float*)carve((size_t)N * HID * 4);   // t1, then t2
  float*          bufB  = (float*)carve((size_t)N * HID * 4);   // out1 -> h1
  float*          bufC  = (float*)carve((size_t)N * H2 * 4);    // out2 -> h2
  unsigned short* w2hi  = (unsigned short*)carve((size_t)HID * H2 * 2);
  unsigned short* w2lo  = (unsigned short*)carve((size_t)HID * H2 * 2);
  float*          gsum  = (float*)carve((size_t)H2 * 4);

  // zero the accumulation targets (graph-capture-safe)
  hipMemsetAsync(bufB, 0, (size_t)N * HID * 4, stream);
  hipMemsetAsync(bufC, 0, (size_t)N * H2 * 4, stream);
  hipMemsetAsync(gsum, 0, (size_t)H2 * 4, stream);

  const int BT = 256;
  auto blocks = [](long long total, int bt) { return (unsigned int)((total + bt - 1) / bt); };

  // normalization
  k_deg_init<<<blocks(N, BT), BT, 0, stream>>>(deg, N);
  k_deg_edges<<<blocks(E, BT), BT, 0, stream>>>(col, deg, E);
  k_dis<<<blocks(N, BT), BT, 0, stream>>>(deg, N);

  // W2 split/transpose
  k_w2pack<<<blocks(HID * H2, BT), BT, 0, stream>>>(W2, w2hi, w2lo);

  // layer 1
  k_gemm1<<<N, BT, 0, stream>>>(x, W1, bufA, N);
  {
    long long work = ((long long)E + N) * (HID / 4);
    k_agg<HID><<<blocks(work, BT), BT, 0, stream>>>(bufA, bufB, row, col, deg, E, N);
  }
  k_bias_relu<HID><<<blocks((long long)N * HID, BT), BT, 0, stream>>>(bufB, b1, (long long)N * HID);

  // layer 2 (WMMA GEMM)
  k_gemm2<<<blocks(N, 16), BT, 0, stream>>>(bufB, w2hi, w2lo, bufA, N);
  {
    long long work = ((long long)E + N) * (H2 / 4);
    k_agg<H2><<<blocks(work, BT), BT, 0, stream>>>(bufA, bufC, row, col, deg, E, N);
  }
  k_bias_relu<H2><<<blocks((long long)N * H2, BT), BT, 0, stream>>>(bufC, b2, (long long)N * H2);

  // pool + head
  k_pool<<<512, 256, 0, stream>>>(bufC, gsum, N);
  k_final<<<1, H2, 0, stream>>>(gsum, Wl, bl, out, N);
}
